// PostAttentionPruner_70291434766422
// MI455X (gfx1250) — compile-verified, matrix operand order checked by reading, not run on hardware
//
#include <hip/hip_runtime.h>
#include <hip/hip_bf16.h>
#include <math.h>

// ---------------------------------------------------------------------------
// PostAttentionPruner for MI455X (gfx1250, wave32, WMMA)
// node MLP: [N,132]@[132,64] -> gelu -> [64,1] -> sigmoid
// edge MLP: [E,54]@[54,48]   -> gelu -> [48,1] -> sigmoid
// Both GEMMs via v_wmma_f32_16x16x32_f16 (f16 in, f32 accum).
// ---------------------------------------------------------------------------

typedef __attribute__((ext_vector_type(16))) _Float16 v16h;
typedef __attribute__((ext_vector_type(8)))  float    v8f;

#define NN      100000      // nodes
#define NE      1600000     // edges
#define HEADS   4
#define DNODE   128
#define DEDGE   48
#define DHID_N  64          // DNODE/2
#define DHID_E  48
#define KIN_N   (DNODE + HEADS)        // 132
#define KIN_E   (DEDGE + HEADS + 2)    // 54
#define KPAD_N  160         // 132 -> 5 chunks of 32
#define KPAD_E  64          // 54  -> 2 chunks of 32

__device__ __forceinline__ float gelu_exact(float x) {
    return 0.5f * x * (1.0f + erff(x * 0.70710678118654752440f));
}
__device__ __forceinline__ float sigm(float x) {
    return 1.0f / (1.0f + expf(-x));
}

// ---------------- scratch init ----------------
__global__ void k_zero(float* __restrict__ sums, unsigned* __restrict__ maxbits) {
    int i = blockIdx.x * blockDim.x + threadIdx.x;
    if (i < NN * HEADS) sums[i] = 0.0f;
    if (i < HEADS) maxbits[i] = 0u;
}

// ---------------- weight transpose + f16 convert (zero K-pad) --------------
__global__ void k_prep_weights(const float* __restrict__ wn1, const float* __restrict__ we1,
                               _Float16* __restrict__ wn1T, _Float16* __restrict__ we1T) {
    for (int i = threadIdx.x; i < DHID_N * KPAD_N; i += blockDim.x) {
        int n = i / KPAD_N, k = i % KPAD_N;
        wn1T[i] = (k < KIN_N) ? (_Float16)wn1[k * DHID_N + n] : (_Float16)0.0f;
    }
    for (int i = threadIdx.x; i < DHID_E * KPAD_E; i += blockDim.x) {
        int n = i / KPAD_E, k = i % KPAD_E;
        we1T[i] = (k < KIN_E) ? (_Float16)we1[k * DHID_E + n] : (_Float16)0.0f;
    }
}

// ---------------- segment-sum of attention onto target nodes ---------------
__global__ void k_scatter(const float* __restrict__ attn, const int* __restrict__ eidx,
                          float* __restrict__ sums) {
    int e = blockIdx.x * blockDim.x + threadIdx.x;
    if (e >= NE) return;
    int t = eidx[NE + e];  // edge_index[1][e]
    float4 a = *(const float4*)(attn + (size_t)e * HEADS);
    atomicAdd(&sums[t * HEADS + 0], a.x);
    atomicAdd(&sums[t * HEADS + 1], a.y);
    atomicAdd(&sums[t * HEADS + 2], a.z);
    atomicAdd(&sums[t * HEADS + 3], a.w);
}

// ---------------- per-head max (sums >= 0, so uint bit-order works) --------
__global__ void k_headmax(const float* __restrict__ sums, unsigned* __restrict__ maxbits) {
    int n = blockIdx.x * blockDim.x + threadIdx.x;
    float h[HEADS] = {0.f, 0.f, 0.f, 0.f};
    if (n < NN) {
        float4 v = *(const float4*)(sums + (size_t)n * HEADS);
        h[0] = v.x; h[1] = v.y; h[2] = v.z; h[3] = v.w;
    }
#pragma unroll
    for (int i = 0; i < HEADS; ++i) {
        float x = h[i];
#pragma unroll
        for (int d = 1; d < 32; d <<= 1) x = fmaxf(x, __shfl_xor(x, d));
        h[i] = x;
    }
    if ((threadIdx.x & 31) == 0) {
#pragma unroll
        for (int i = 0; i < HEADS; ++i)
            atomicMax(&maxbits[i], __float_as_uint(h[i]));
    }
}

// ---------------- node gate MLP: one wave per 16-node tile -----------------
__global__ __launch_bounds__(256) void k_node_gate(
    const float* __restrict__ nodef, const float* __restrict__ sums,
    const unsigned* __restrict__ maxbits,
    const _Float16* __restrict__ wn1T, const float* __restrict__ bn1,
    const float* __restrict__ wn2, const float* __restrict__ bn2,
    float* __restrict__ node_gates)
{
    const int lane = threadIdx.x & 31;
    const int tile = (blockIdx.x * blockDim.x + threadIdx.x) >> 5;  // wave-uniform
    if (tile * 16 >= NN) return;                                    // uniform exit
    const int half = lane >> 4;   // K-half for A frags
    const int m    = lane & 15;   // row (A) / hidden column (B,C)
    const int row  = tile * 16 + m;

    float invmax[HEADS];
#pragma unroll
    for (int h = 0; h < HEADS; ++h)
        invmax[h] = 1.0f / (__uint_as_float(maxbits[h]) + 1e-10f);

    v8f acc[4];
#pragma unroll
    for (int nt = 0; nt < 4; ++nt) acc[nt] = (v8f){0,0,0,0,0,0,0,0};

#pragma unroll
    for (int kc = 0; kc < 5; ++kc) {
        const int k0 = kc * 32 + half * 16;   // lane's 16 contiguous K values
        v16h a;
        if (k0 < DNODE) {                      // uniform for kc<4
            const float4* p = (const float4*)(nodef + (size_t)row * DNODE + k0);
#pragma unroll
            for (int q = 0; q < 4; ++q) {
                float4 f = p[q];
                a[q*4+0] = (_Float16)f.x; a[q*4+1] = (_Float16)f.y;
                a[q*4+2] = (_Float16)f.z; a[q*4+3] = (_Float16)f.w;
            }
        } else {                               // kc==4: K tail = normalized attn + zeros
#pragma unroll
            for (int q = 0; q < 16; ++q) a[q] = (_Float16)0.0f;
            if (k0 == DNODE) {                 // lanes 0..15 only
                float4 s = *(const float4*)(sums + (size_t)row * HEADS);
                a[0] = (_Float16)(s.x * invmax[0]); a[1] = (_Float16)(s.y * invmax[1]);
                a[2] = (_Float16)(s.z * invmax[2]); a[3] = (_Float16)(s.w * invmax[3]);
            }
        }
#pragma unroll
        for (int nt = 0; nt < 4; ++nt) {
            const v16h b = *(const v16h*)(wn1T + (size_t)(nt * 16 + m) * KPAD_N + k0);
            acc[nt] = __builtin_amdgcn_wmma_f32_16x16x32_f16(
                false, a, false, b, (short)0, acc[nt], false, false);
        }
    }

    // Layer 2: C layout -> lane holds column n=m, VGPR r holds row (half*8+r).
    float partial[8];
#pragma unroll
    for (int r = 0; r < 8; ++r) partial[r] = 0.0f;
#pragma unroll
    for (int nt = 0; nt < 4; ++nt) {
        const int n = nt * 16 + m;
        const float b1 = bn1[n];
        const float w2 = wn2[n];
#pragma unroll
        for (int r = 0; r < 8; ++r)
            partial[r] += gelu_exact(acc[nt][r] + b1) * w2;
    }
    const float b2 = bn2[0];
#pragma unroll
    for (int r = 0; r < 8; ++r) {
        float v = partial[r];
        v += __shfl_xor(v, 1);
        v += __shfl_xor(v, 2);
        v += __shfl_xor(v, 4);
        v += __shfl_xor(v, 8);   // reduce over 16-lane half (columns)
        partial[r] = v;
    }
    if (m == 0) {
        const int base = tile * 16 + half * 8;
#pragma unroll
        for (int r = 0; r < 8; ++r)
            node_gates[base + r] = sigm(partial[r] + b2);
    }
}

// ---------------- edge gate MLP: one wave per 16-edge tile -----------------
__global__ __launch_bounds__(256) void k_edge_gate(
    const float* __restrict__ edgef, const float* __restrict__ attn,
    const int* __restrict__ eidx, const float* __restrict__ node_gates,
    const _Float16* __restrict__ we1T, const float* __restrict__ be1,
    const float* __restrict__ we2, const float* __restrict__ be2,
    float* __restrict__ edge_gates)
{
    const int lane = threadIdx.x & 31;
    const int tile = (blockIdx.x * blockDim.x + threadIdx.x) >> 5;  // wave-uniform
    if (tile * 16 >= NE) return;
    const int half = lane >> 4;
    const int m    = lane & 15;
    const int e    = tile * 16 + m;

    v8f acc[3];
#pragma unroll
    for (int nt = 0; nt < 3; ++nt) acc[nt] = (v8f){0,0,0,0,0,0,0,0};

#pragma unroll
    for (int kc = 0; kc < 2; ++kc) {
        const int k0 = kc * 32 + half * 16;   // 0,16,32,48
        v16h a;
        if (k0 < DEDGE) {
            const float4* p = (const float4*)(edgef + (size_t)e * DEDGE + k0);
#pragma unroll
            for (int q = 0; q < 4; ++q) {
                float4 f = p[q];
                a[q*4+0] = (_Float16)f.x; a[q*4+1] = (_Float16)f.y;
                a[q*4+2] = (_Float16)f.z; a[q*4+3] = (_Float16)f.w;
            }
        } else {  // k0==48: [attn(4), src_gate, tgt_gate, zeros(10)]
#pragma unroll
            for (int q = 0; q < 16; ++q) a[q] = (_Float16)0.0f;
            float4 s = *(const float4*)(attn + (size_t)e * HEADS);
            const int src = eidx[e];
            const int tgt = eidx[NE + e];
            a[0] = (_Float16)s.x; a[1] = (_Float16)s.y;
            a[2] = (_Float16)s.z; a[3] = (_Float16)s.w;
            a[4] = (_Float16)node_gates[src];
            a[5] = (_Float16)node_gates[tgt];
        }
#pragma unroll
        for (int nt = 0; nt < 3; ++nt) {
            const v16h b = *(const v16h*)(we1T + (size_t)(nt * 16 + m) * KPAD_E + k0);
            acc[nt] = __builtin_amdgcn_wmma_f32_16x16x32_f16(
                false, a, false, b, (short)0, acc[nt], false, false);
        }
    }

    float partial[8];
#pragma unroll
    for (int r = 0; r < 8; ++r) partial[r] = 0.0f;
#pragma unroll
    for (int nt = 0; nt < 3; ++nt) {
        const int n = nt * 16 + m;
        const float b1 = be1[n];
        const float w2 = we2[n];
#pragma unroll
        for (int r = 0; r < 8; ++r)
            partial[r] += gelu_exact(acc[nt][r] + b1) * w2;
    }
    const float b2 = be2[0];
#pragma unroll
    for (int r = 0; r < 8; ++r) {
        float v = partial[r];
        v += __shfl_xor(v, 1);
        v += __shfl_xor(v, 2);
        v += __shfl_xor(v, 4);
        v += __shfl_xor(v, 8);
        partial[r] = v;
    }
    if (m == 0) {
        const int base = tile * 16 + half * 8;
#pragma unroll
        for (int r = 0; r < 8; ++r)
            edge_gates[base + r] = sigm(partial[r] + b2);
    }
}

// ---------------------------------------------------------------------------
extern "C" void kernel_launch(void* const* d_in, const int* in_sizes, int n_in,
                              void* d_out, int out_size, void* d_ws, size_t ws_size,
                              hipStream_t stream) {
    const float* nodef = (const float*)d_in[0];
    const float* edgef = (const float*)d_in[1];
    const float* attn  = (const float*)d_in[2];
    const int*   eidx  = (const int*)d_in[3];
    const float* wn1   = (const float*)d_in[4];
    const float* bn1   = (const float*)d_in[5];
    const float* wn2   = (const float*)d_in[6];
    const float* bn2   = (const float*)d_in[7];
    const float* we1   = (const float*)d_in[8];
    const float* be1   = (const float*)d_in[9];
    const float* we2   = (const float*)d_in[10];
    const float* be2   = (const float*)d_in[11];

    float* node_gates = (float*)d_out;        // [NN]
    float* edge_gates = node_gates + NN;      // [NE]

    // workspace carve (64B-align the f16 weight region for v16h loads)
    char* ws = (char*)d_ws;
    float*    sums    = (float*)ws;                                   // NN*4 f32
    unsigned* maxbits = (unsigned*)(ws + (size_t)NN * HEADS * 4);     // 4 u32
    size_t woff = ((size_t)NN * HEADS * 4 + HEADS * 4 + 63) & ~(size_t)63;
    _Float16* wn1T = (_Float16*)(ws + woff);                          // 64*160 f16
    _Float16* we1T = wn1T + (size_t)DHID_N * KPAD_N;                  // 48*64  f16

    k_zero<<<(NN * HEADS + 255) / 256, 256, 0, stream>>>(sums, maxbits);
    k_prep_weights<<<1, 256, 0, stream>>>(wn1, we1, wn1T, we1T);
    k_scatter<<<(NE + 255) / 256, 256, 0, stream>>>(attn, eidx, sums);
    k_headmax<<<(NN + 255) / 256, 256, 0, stream>>>(sums, maxbits);

    const int nodeTiles = (NN + 15) / 16;   // 6250 waves
    const int nodeBlocks = (nodeTiles * 32 + 255) / 256;
    k_node_gate<<<nodeBlocks, 256, 0, stream>>>(nodef, sums, maxbits,
                                                wn1T, bn1, wn2, bn2, node_gates);

    const int edgeTiles = (NE + 15) / 16;   // 100000 waves
    const int edgeBlocks = (edgeTiles * 32 + 255) / 256;
    k_edge_gate<<<edgeBlocks, 256, 0, stream>>>(edgef, attn, eidx, node_gates,
                                                we1T, be1, we2, be2, edge_gates);
}
